// MLP_42039139893450
// MI455X (gfx1250) — compile-verified, matrix operand order checked by reading, not run on hardware
//
#include <hip/hip_runtime.h>
#include <hip/hip_bf16.h>

// Workspace layout (requires >= 160 MiB):
//   [0, 64MiB)        wq_gu  bf16 [16384 x 2048]
//   [64MiB, 96MiB)    wq_p   bf16 [2048 x 8192]
//   [96MiB, 160MiB)   h      bf16 [4096 x 8192]  (rms-normalized in place)

typedef __attribute__((ext_vector_type(16))) __bf16 v16bf;
typedef __attribute__((ext_vector_type(8)))  __bf16 v8bf;
typedef __attribute__((ext_vector_type(8)))  float  v8f;

#define GROUP 64

constexpr int BM = 128, BN = 64, BK = 64;
constexpr int LD = BK + 8;        // pad: row stride 144B breaks 64-bank conflicts

// ---------------- CDNA5 async memory->LDS (ASYNCcnt-tracked, no VGPR data) ----
__device__ __forceinline__ void async_b128(unsigned lds_off, const __bf16* g) {
  // GLOBAL_LOAD_ASYNC_TO_LDS_B128: vdst = LDS byte offset, vaddr = 64-bit addr
  asm volatile("global_load_async_to_lds_b128 %0, %1, off"
               :: "v"(lds_off), "v"(g) : "memory");
}

__device__ __forceinline__ void wait_async0() {
#if __has_builtin(__builtin_amdgcn_s_wait_asynccnt)
  __builtin_amdgcn_s_wait_asynccnt(0);
#else
  asm volatile("s_wait_asynccnt 0x0" ::: "memory");
#endif
}

// ---------------- per-group ternary quantization (group 64 == one wave32 x 2) -
__global__ void __launch_bounds__(256) quant_ternary(const float* __restrict__ w,
                                                     __bf16* __restrict__ wq,
                                                     int n) {
  int idx  = blockIdx.x * blockDim.x + threadIdx.x;
  int base = idx * 2;
  if (base >= n) return;
  float f0 = w[base + 0];
  float f1 = w[base + 1];
  __bf16 b0 = (__bf16)f0;                 // reference quantizes in bf16
  __bf16 b1 = (__bf16)f1;
  float s = fabsf((float)b0) + fabsf((float)b1);
#pragma unroll
  for (int off = 16; off; off >>= 1) s += __shfl_xor(s, off, 32);
  float scale = (float)(__bf16)(s * (1.0f / GROUP));   // mean rounded to bf16
  scale = fmaxf(scale, 1e-8f);
  float inv = 1.0f / scale;
  float q0 = fminf(fmaxf(rintf((float)b0 * inv), -1.0f), 1.0f);
  float q1 = fminf(fmaxf(rintf((float)b1 * inv), -1.0f), 1.0f);
  wq[base + 0] = (__bf16)(q0 * scale);
  wq[base + 1] = (__bf16)(q1 * scale);
}

// ---------------- WMMA fragment load from an LDS tile stored row-major --------
// 16-bit A/B layout (ISA 7.12.2): lane&15 = row, lane>>4 selects K-half;
// each lane holds K {0..7,16..23} or {8..15,24..31} => two b128 LDS loads.
__device__ __forceinline__ v16bf ld_frag(const __bf16* p) {
  int lane = threadIdx.x & 31;
  const __bf16* q = p + (lane & 15) * LD + ((lane >> 4) << 3);
  v8bf a0 = *(const v8bf*)q;
  v8bf a1 = *(const v8bf*)(q + 16);
  return __builtin_shufflevector(a0, a1, 0,1,2,3,4,5,6,7,8,9,10,11,12,13,14,15);
}

__device__ __forceinline__ v8f wmma_bf16(v16bf a, v16bf b, v8f c) {
  return __builtin_amdgcn_wmma_f32_16x16x32_bf16(false, a, false, b,
                                                 (short)0, c, false, false);
}

// ---------------- tiled bf16 WMMA GEMM: C[M,N] = A[M,K] * B[N,K]^T -----------
// FUSE=true: also accumulates up = A * B[n+upRowOff,:]^T and writes
// silu(gate)*up. Tile: 128(M) x 64(N) x 64(K); 8 waves = 4(M) x 2(N),
// each wave owns a 32x32 patch = 2x2 WMMA tiles. Double-buffered async LDS.
template <bool FUSE>
__global__ void __launch_bounds__(256) gemm_wmma(const __bf16* __restrict__ A,
                                                 const __bf16* __restrict__ B,
                                                 __bf16* __restrict__ C,
                                                 int K, int lda, int ldb, int ldc,
                                                 int upRowOff) {
  __shared__ __align__(16) __bf16 sA [2][BM * LD];
  __shared__ __align__(16) __bf16 sBg[2][BN * LD];
  __shared__ __align__(16) __bf16 sBu[2][BN * LD];   // unused when !FUSE

  const int mBase = blockIdx.y * BM;
  const int nBase = blockIdx.x * BN;
  const int wave  = threadIdx.x >> 5;
  const int wm    = wave & 3;                // 4 waves along M
  const int wn    = wave >> 2;               // 2 waves along N

  const int t     = threadIdx.x;
  const int chunk = (t & 7) * 8;             // 8 bf16 (16B) per chunk
  const int row   = t >> 3;                  // 32 rows per sweep

  v8f accG[2][2] = {};
  v8f accU[2][2] = {};

  // ---- branch-free async staging of one K-tile into LDS buffer `buf` ----
  auto stage = [&](int kt, int buf) {
#pragma unroll
    for (int s = 0; s < 4; ++s) {            // A tile: 128 x 64
      int r = row + s * 32;
      async_b128((unsigned)(uintptr_t)&sA[buf][r * LD + chunk],
                 A + (size_t)(mBase + r) * lda + kt + chunk);
    }
#pragma unroll
    for (int s = 0; s < 2; ++s) {            // weight tile(s): 64 x 64
      int r = row + s * 32;
      const __bf16* gp = B + (size_t)(nBase + r) * ldb + kt + chunk;
      async_b128((unsigned)(uintptr_t)&sBg[buf][r * LD + chunk], gp);
      if (FUSE)
        async_b128((unsigned)(uintptr_t)&sBu[buf][r * LD + chunk],
                   gp + (size_t)upRowOff * ldb);
    }
  };

  const int nT = K / BK;
  stage(0, 0);
  wait_async0();
  __syncthreads();

  for (int it = 0; it < nT; ++it) {
    const int cur = it & 1;
    const int nxt = cur ^ 1;
    const bool more = (it + 1) < nT;

    // issue async loads for the next tile while we compute on the current one
    if (more) stage((it + 1) * BK, nxt);

    // branch-free L2 prefetch two tiles ahead (bf16 weights live in 192MB L2)
    {
      int kpf = (it + 2) * BK; kpf = kpf > K - BK ? K - BK : kpf;
      __builtin_prefetch(A + (size_t)(mBase + row) * lda + kpf + chunk, 0, 2);
      __builtin_prefetch(B + (size_t)(nBase + (row & 63)) * ldb + kpf + chunk, 0, 2);
    }

#pragma unroll
    for (int kk = 0; kk < BK; kk += 32) {
      v16bf a0 = ld_frag(&sA[cur][(wm * 32 +  0) * LD + kk]);
      v16bf a1 = ld_frag(&sA[cur][(wm * 32 + 16) * LD + kk]);
      v16bf g0 = ld_frag(&sBg[cur][(wn * 32 +  0) * LD + kk]);
      v16bf g1 = ld_frag(&sBg[cur][(wn * 32 + 16) * LD + kk]);
      accG[0][0] = wmma_bf16(a0, g0, accG[0][0]);
      accG[0][1] = wmma_bf16(a0, g1, accG[0][1]);
      accG[1][0] = wmma_bf16(a1, g0, accG[1][0]);
      accG[1][1] = wmma_bf16(a1, g1, accG[1][1]);
      if (FUSE) {
        v16bf u0 = ld_frag(&sBu[cur][(wn * 32 +  0) * LD + kk]);
        v16bf u1 = ld_frag(&sBu[cur][(wn * 32 + 16) * LD + kk]);
        accU[0][0] = wmma_bf16(a0, u0, accU[0][0]);
        accU[0][1] = wmma_bf16(a0, u1, accU[0][1]);
        accU[1][0] = wmma_bf16(a1, u0, accU[1][0]);
        accU[1][1] = wmma_bf16(a1, u1, accU[1][1]);
      }
    }

    if (more) wait_async0();   // our async transfers into buf `nxt` landed
    __syncthreads();           // all waves done: next tile visible, cur reusable
  }

  // ---- epilogue: C/D layout (ISA 7.12.2): VGPR i -> M = i + 8*(lane>>4),
  //      N = lane&15
  const int lane = threadIdx.x & 31;
  const int mo   = (lane >> 4) * 8;
  const int no   = lane & 15;
#pragma unroll
  for (int tm = 0; tm < 2; ++tm)
#pragma unroll
    for (int tn = 0; tn < 2; ++tn) {
      v8f g = accG[tm][tn];
      v8f u = accU[tm][tn];
      size_t rbase = (size_t)(mBase + wm * 32 + tm * 16 + mo);
      int    col   = nBase + wn * 32 + tn * 16 + no;
#pragma unroll
      for (int i = 0; i < 8; ++i) {
        float v = g[i];
        if (FUSE) v = (v / (1.0f + __expf(-v))) * u[i];  // silu(g)*u
        C[(rbase + i) * (size_t)ldc + col] = (__bf16)v;
      }
    }
}

// ---------------- in-place RMS norm over rows of length Hdim ------------------
__global__ void __launch_bounds__(256) rmsnorm_inplace(__bf16* __restrict__ h,
                                                       int Hdim) {
  __shared__ float red[8];
  const size_t base = (size_t)blockIdx.x * Hdim;
  const int t = threadIdx.x;
  v8bf v[4];
  float ss = 0.0f;
#pragma unroll
  for (int j = 0; j < 4; ++j) {
    v[j] = *(const v8bf*)(h + base + t * 32 + j * 8);
#pragma unroll
    for (int i = 0; i < 8; ++i) { float f = (float)v[j][i]; ss += f * f; }
  }
#pragma unroll
  for (int off = 16; off; off >>= 1) ss += __shfl_xor(ss, off, 32);
  const int wv = t >> 5, lane = t & 31;
  if (lane == 0) red[wv] = ss;
  __syncthreads();
  float tot = red[0] + red[1] + red[2] + red[3] +
              red[4] + red[5] + red[6] + red[7];
  // eps = bf16 machine eps = 2^-7 (torch F.rms_norm default for bf16)
  float inv = rsqrtf(tot / (float)Hdim + 0.0078125f);
#pragma unroll
  for (int j = 0; j < 4; ++j) {
    v8bf o;
#pragma unroll
    for (int i = 0; i < 8; ++i) o[i] = (__bf16)((float)v[j][i] * inv);
    *(v8bf*)(h + base + t * 32 + j * 8) = o;
  }
}

extern "C" void kernel_launch(void* const* d_in, const int* in_sizes, int n_in,
                              void* d_out, int out_size, void* d_ws, size_t ws_size,
                              hipStream_t stream) {
  const __bf16* x    = (const __bf16*)d_in[0];      // [4096, 2048] bf16
  const float*  w_gu = (const float*)d_in[1];       // [16384, 2048] f32
  const float*  w_p  = (const float*)d_in[2];       // [2048, 8192] f32
  __bf16* out = (__bf16*)d_out;                     // [4096, 2048] bf16

  const int Dm = 2048, H = 4 * Dm, M = 4096;
  const size_t n_gu = (size_t)2 * H * Dm;           // 33,554,432
  const size_t n_p  = (size_t)Dm * H;               // 16,777,216

  char* ws = (char*)d_ws;
  __bf16* wq_gu = (__bf16*)ws;  ws += n_gu * sizeof(__bf16);
  __bf16* wq_p  = (__bf16*)ws;  ws += n_p  * sizeof(__bf16);
  __bf16* hbuf  = (__bf16*)ws;                      // [4096, 8192] bf16

  // 1) quantize weights (groups of 64 along K, contiguous)
  quant_ternary<<<dim3((unsigned)(n_gu / 512)), 256, 0, stream>>>(w_gu, wq_gu, (int)n_gu);
  quant_ternary<<<dim3((unsigned)(n_p  / 512)), 256, 0, stream>>>(w_p,  wq_p,  (int)n_p);

  // 2) h = silu(x @ wq_gu[0:H]^T) * (x @ wq_gu[H:2H]^T)
  gemm_wmma<true><<<dim3(H / 64, M / 128), 256, 0, stream>>>(
      x, wq_gu, hbuf, /*K=*/Dm, /*lda=*/Dm, /*ldb=*/Dm, /*ldc=*/H, /*upRowOff=*/H);

  // 3) h <- rms_norm(h), in place
  rmsnorm_inplace<<<dim3(M), 256, 0, stream>>>(hbuf, H);

  // 4) out = h @ wq_p^T
  gemm_wmma<false><<<dim3(Dm / 64, M / 128), 256, 0, stream>>>(
      hbuf, wq_p, out, /*K=*/H, /*lda=*/H, /*ldb=*/H, /*ldc=*/Dm, /*upRowOff=*/0);
}
